// GGNNClassifierFeatsNoEmb_66254165508838
// MI455X (gfx1250) — matrix operand order repaired and checked by reading, not compiled, not run on hardware
//
#include <hip/hip_runtime.h>
#include <hip/hip_bf16.h>

#define NNODES   50000
#define NPAD     50048      /* 64 * 782, padded row count for GEMM tiling */
#define NEDGES   500000
#define CDIM     128
#define C3       384
#define NTYPES   32
#define DIMTOK   94
#define NSTEPS   3
#define NBLOCKS  2
#define NET      3
#define LN_EPS_  1e-5f

typedef __attribute__((ext_vector_type(16))) __bf16 v16bf;
typedef __attribute__((ext_vector_type(8)))  float  v8f;

// ---------------------------------------------------------------------------
// WMMA GEMM:  D[NPAD x ncols] = A[NPAD x 128] (f32 -> bf16 in-register)
//                             x Bsw (bf16, pre-swizzled to fragment layout)
//                             + bias[ncols], optional ReLU
// Block = 128 threads (4 waves). Each wave: 4 row-tiles x 2 col-tiles (64x32),
// so each A fragment feeds 2 WMMAs and each B fragment feeds 4 WMMAs:
// 32 WMMAs per wave, per-WMMA cost = 2 x b128 A-load + 4 x cvt_pk + 1/4 B-load.
// Grid = (NPAD/64, ncols/128). K = 128.
// Fragment layouts per CDNA5 ISA 7.12.2 (wave32):
//   A 16x32 bf16 : lane l (l&15 = row M); elem e: K = kb*32 + (e<8?e:e+8) + 8*(l>>4)
//   B 32x16 bf16 : lane l (l&15 = col N); elem e: K = kb*32 + (e<8?e:e+8) + 8*(l>>4)
//     -> pre-swizzled: Bsw[((ct*4 + kb)*32 + lane)*16 + e]  (contiguous per lane)
//   C/D 16x16 f32: lane l (l&15 = col N); vgpr r: row M = r + 8*(l>>4)
// ---------------------------------------------------------------------------
__global__ void __launch_bounds__(128)
gemm_wmma_bf16(const float* __restrict__ A, const __bf16* __restrict__ Bsw,
               const float* __restrict__ bias, float* __restrict__ D,
               int ncols, int do_relu)
{
    const int lane    = threadIdx.x & 31;
    const int wave    = threadIdx.x >> 5;
    const int l15     = lane & 15;
    const int half    = lane >> 4;
    const int rowbase = blockIdx.x * 64;
    const int ct0     = (blockIdx.y * 4 + wave) * 2;     // two col tiles
    const int col0    = ct0 * 16 + l15;

    v8f acc[4][2];
#pragma unroll
    for (int c = 0; c < 2; ++c) {
        const float binit = bias ? bias[col0 + c * 16] : 0.0f;
#pragma unroll
        for (int r = 0; r < 4; ++r)
#pragma unroll
            for (int i = 0; i < 8; ++i) acc[r][c][i] = binit;
    }

    const float*  Abase = A + (size_t)(rowbase + l15) * CDIM + half * 8;
    const __bf16* Bbase = Bsw + (size_t)ct0 * 4 * 512 + lane * 16;

#pragma unroll
    for (int kb = 0; kb < 4; ++kb) {
        const v16bf b0 = *(const v16bf*)(Bbase + kb * 512);
        const v16bf b1 = *(const v16bf*)(Bbase + (4 + kb) * 512);
#pragma unroll
        for (int r = 0; r < 4; ++r) {
            const float* Ap = Abase + (size_t)r * 16 * CDIM + kb * 32;
            const float4 f0 = *(const float4*)(Ap + 0);
            const float4 f1 = *(const float4*)(Ap + 4);
            const float4 f2 = *(const float4*)(Ap + 16);
            const float4 f3 = *(const float4*)(Ap + 20);
            v16bf a;
            a[0]  = (__bf16)f0.x; a[1]  = (__bf16)f0.y;
            a[2]  = (__bf16)f0.z; a[3]  = (__bf16)f0.w;
            a[4]  = (__bf16)f1.x; a[5]  = (__bf16)f1.y;
            a[6]  = (__bf16)f1.z; a[7]  = (__bf16)f1.w;
            a[8]  = (__bf16)f2.x; a[9]  = (__bf16)f2.y;
            a[10] = (__bf16)f2.z; a[11] = (__bf16)f2.w;
            a[12] = (__bf16)f3.x; a[13] = (__bf16)f3.y;
            a[14] = (__bf16)f3.z; a[15] = (__bf16)f3.w;
            acc[r][0] = __builtin_amdgcn_wmma_f32_16x16x32_bf16(
                false, a, false, b0, (short)0, acc[r][0], false, false);
            acc[r][1] = __builtin_amdgcn_wmma_f32_16x16x32_bf16(
                false, a, false, b1, (short)0, acc[r][1], false, false);
        }
    }

#pragma unroll
    for (int r = 0; r < 4; ++r) {
#pragma unroll
        for (int c = 0; c < 2; ++c) {
#pragma unroll
            for (int i = 0; i < 8; ++i) {
                const int row = rowbase + r * 16 + i + 8 * half;
                float v = acc[r][c][i];
                if (do_relu) v = v > 0.0f ? v : 0.0f;
                D[(size_t)row * ncols + col0 + c * 16] = v;
            }
        }
    }
}

// ---------------------------------------------------------------------------
// Repack fp32 weights into bf16 WMMA-B fragment order.
// dst[(((mat*(ncols/16) + ct)*4 + kb)*32 + lane)*16 + e] =
//   src[mat][k][col]            (transpose==0, src per-mat is K x ncols)
//   src[mat][col][k]            (transpose==1, src per-mat is ncols x K)
// with k = kb*32 + (lane>>4)*8 + (e<8?e:e+8), col = ct*16 + (lane&15), K = 128.
// ---------------------------------------------------------------------------
__global__ void __launch_bounds__(256)
repack_b(const float* __restrict__ src, __bf16* __restrict__ dst,
         int ncols, int transpose, int total)
{
    const int idx = blockIdx.x * blockDim.x + threadIdx.x;
    if (idx >= total) return;
    const int e    = idx & 15;
    const int lane = (idx >> 4) & 31;
    const int kb   = (idx >> 9) & 3;
    const int rest = idx >> 11;            // mat*(ncols/16) + ct
    const int nct  = ncols >> 4;
    const int ct   = rest % nct;
    const int mat  = rest / nct;
    const int k    = kb * 32 + (lane >> 4) * 8 + (e < 8 ? e : e + 8);
    const int col  = ct * 16 + (lane & 15);
    const size_t matoff = (size_t)mat * CDIM * ncols;
    const size_t s = matoff + (transpose ? ((size_t)col * CDIM + k)
                                         : ((size_t)k * ncols + col));
    dst[idx] = (__bf16)src[s];
}

// ---------------------------------------------------------------------------
// Edge scatter-add: one wave per edge; 4 floats per lane; fp32 HW atomics.
// ---------------------------------------------------------------------------
__global__ void __launch_bounds__(256)
scatter_edges(const float* __restrict__ x, const int* __restrict__ esrc,
              const int* __restrict__ edst, const int* __restrict__ etype,
              int t, float* __restrict__ agg)
{
    const int e = blockIdx.x * 8 + (threadIdx.x >> 5);
    if (e >= NEDGES) return;
    if (etype[e] != t) return;
    const int lane = threadIdx.x & 31;
    const float4 v = *(const float4*)(x + (size_t)esrc[e] * CDIM + lane * 4);
    float* ad = agg + (size_t)edst[e] * CDIM + lane * 4;
    unsafeAtomicAdd(ad + 0, v.x);
    unsafeAtomicAdd(ad + 1, v.y);
    unsafeAtomicAdd(ad + 2, v.z);
    unsafeAtomicAdd(ad + 3, v.w);
}

// ---------------------------------------------------------------------------
// GRU gate combine (r,z,n ordering per torch GRUCell), updates m in place.
// ---------------------------------------------------------------------------
__global__ void __launch_bounds__(256)
gru_eltwise(const float* __restrict__ gi, const float* __restrict__ gh,
            float* __restrict__ m)
{
    const int idx = blockIdx.x * blockDim.x + threadIdx.x;
    if (idx >= NPAD * CDIM) return;
    const int n = idx >> 7;
    const int c = idx & 127;
    const float* gin_ = gi + (size_t)n * C3;
    const float* ghn_ = gh + (size_t)n * C3;
    float r  = gin_[c]            + ghn_[c];
    float z  = gin_[CDIM + c]     + ghn_[CDIM + c];
    float nn = gin_[2 * CDIM + c];
    float hn = ghn_[2 * CDIM + c];
    r = 1.0f / (1.0f + __expf(-r));
    z = 1.0f / (1.0f + __expf(-z));
    nn = tanhf(nn + r * hn);
    const float h = m[idx];
    m[idx] = (1.0f - z) * nn + z * h;
}

// ---------------------------------------------------------------------------
// LayerNorm(h + total)*g + b then ReLU, written back into h. One wave/node.
// ---------------------------------------------------------------------------
__global__ void __launch_bounds__(256)
ln_relu(const float* __restrict__ total, const float* __restrict__ g,
        const float* __restrict__ bta, float* __restrict__ h)
{
    const int node = blockIdx.x * 8 + (threadIdx.x >> 5);
    if (node >= NPAD) return;
    const int lane = threadIdx.x & 31;
    float* hp = h + (size_t)node * CDIM + lane * 4;
    const float4 hv = *(const float4*)hp;
    const float4 tv = *(const float4*)(total + (size_t)node * CDIM + lane * 4);
    float v0 = hv.x + tv.x, v1 = hv.y + tv.y, v2 = hv.z + tv.z, v3 = hv.w + tv.w;
    float s  = v0 + v1 + v2 + v3;
    float sq = v0 * v0 + v1 * v1 + v2 * v2 + v3 * v3;
#pragma unroll
    for (int off = 16; off >= 1; off >>= 1) {
        s  += __shfl_xor(s, off, 32);
        sq += __shfl_xor(sq, off, 32);
    }
    const float mu  = s * (1.0f / CDIM);
    const float var = sq * (1.0f / CDIM) - mu * mu;
    const float rs  = rsqrtf(var + LN_EPS_);
    const int c = lane * 4;
    float o0 = (v0 - mu) * rs * g[c + 0] + bta[c + 0];
    float o1 = (v1 - mu) * rs * g[c + 1] + bta[c + 1];
    float o2 = (v2 - mu) * rs * g[c + 2] + bta[c + 2];
    float o3 = (v3 - mu) * rs * g[c + 3] + bta[c + 3];
    float4 o;
    o.x = o0 > 0.0f ? o0 : 0.0f;
    o.y = o1 > 0.0f ? o1 : 0.0f;
    o.z = o2 > 0.0f ? o2 : 0.0f;
    o.w = o3 > 0.0f ? o3 : 0.0f;
    *(float4*)hp = o;
}

// ---------------------------------------------------------------------------
// Final head: out[n, j] = z[n,:] . w2[j,:] + b2[j], j in {0,1}. One wave/node.
// ---------------------------------------------------------------------------
__global__ void __launch_bounds__(256)
head2_kernel(const float* __restrict__ z, const float* __restrict__ w2,
             const float* __restrict__ b2, float* __restrict__ out)
{
    const int node = blockIdx.x * 8 + (threadIdx.x >> 5);
    if (node >= NNODES) return;
    const int lane = threadIdx.x & 31;
    const float4 zv = *(const float4*)(z + (size_t)node * CDIM + lane * 4);
    const float4 w0 = *(const float4*)(w2 + lane * 4);
    const float4 w1 = *(const float4*)(w2 + CDIM + lane * 4);
    float d0 = zv.x * w0.x + zv.y * w0.y + zv.z * w0.z + zv.w * w0.w;
    float d1 = zv.x * w1.x + zv.y * w1.y + zv.z * w1.z + zv.w * w1.w;
#pragma unroll
    for (int off = 16; off >= 1; off >>= 1) {
        d0 += __shfl_xor(d0, off, 32);
        d1 += __shfl_xor(d1, off, 32);
    }
    if (lane == 0) {
        out[(size_t)node * 2 + 0] = d0 + b2[0];
        out[(size_t)node * 2 + 1] = d1 + b2[1];
    }
}

// ---------------------------------------------------------------------------
// Feature build over padded rows: one-hot(type) | one-hot(tok) | x_small;
// rows >= NNODES are zero so all padded GEMM inputs are well-defined.
// ---------------------------------------------------------------------------
__global__ void __launch_bounds__(256)
build_features(const int* __restrict__ x_type, const int* __restrict__ x_tok,
               const float* __restrict__ x_small, float* __restrict__ h)
{
    const int idx = blockIdx.x * blockDim.x + threadIdx.x;
    if (idx >= NPAD * CDIM) return;
    const int n = idx >> 7;
    const int c = idx & 127;
    float v = 0.0f;
    if (n < NNODES) {
        if (c < NTYPES) {
            v = (x_type[n] == c) ? 1.0f : 0.0f;
        } else if (c < NTYPES + DIMTOK) {
            int tok = x_tok[n];
            tok = tok < 0 ? 0 : (tok > DIMTOK - 1 ? DIMTOK - 1 : tok);
            v = (tok == c - NTYPES) ? 1.0f : 0.0f;
        } else {
            v = x_small[(size_t)n * 2 + (c - NTYPES - DIMTOK)];
        }
    }
    h[idx] = v;
}

// ---------------------------------------------------------------------------
// Utility kernels
// ---------------------------------------------------------------------------
__global__ void __launch_bounds__(256)
zero_f32(float* __restrict__ p, int n)
{
    const int idx = blockIdx.x * blockDim.x + threadIdx.x;
    if (idx < n) p[idx] = 0.0f;
}

__global__ void __launch_bounds__(256)
copy_f32(float* __restrict__ dst, const float* __restrict__ src, int n)
{
    const int idx = blockIdx.x * blockDim.x + threadIdx.x;
    if (idx < n) dst[idx] = src[idx];
}

__global__ void __launch_bounds__(256)
add_f32(float* __restrict__ dst, const float* __restrict__ src, int n)
{
    const int idx = blockIdx.x * blockDim.x + threadIdx.x;
    if (idx < n) dst[idx] += src[idx];
}

// ---------------------------------------------------------------------------
extern "C" void kernel_launch(void* const* d_in, const int* in_sizes, int n_in,
                              void* d_out, int out_size, void* d_ws, size_t ws_size,
                              hipStream_t stream)
{
    const int*   x_type  = (const int*)d_in[0];
    const int*   x_tok   = (const int*)d_in[1];
    const float* x_small = (const float*)d_in[2];
    const int*   eidx    = (const int*)d_in[3];
    const int*   esrc    = eidx;
    const int*   edst    = eidx + NEDGES;
    const int*   etype   = (const int*)d_in[4];
    const float* conv_w  = (const float*)d_in[5];
    const float* gwih    = (const float*)d_in[6];
    const float* gwhh    = (const float*)d_in[7];
    const float* gbih    = (const float*)d_in[8];
    const float* gbhh    = (const float*)d_in[9];
    const float* ln_g    = (const float*)d_in[10];
    const float* ln_b    = (const float*)d_in[11];
    const float* hw1     = (const float*)d_in[12];
    const float* hb1     = (const float*)d_in[13];
    const float* hw2     = (const float*)d_in[14];
    const float* hb2     = (const float*)d_in[15];

    // -------- workspace carve-out --------
    char*  ws  = (char*)d_ws;
    size_t off = 0;
    auto carve = [&](size_t bytes) -> void* {
        void* p = ws + off;
        off += bytes;
        off = (off + 255) & ~(size_t)255;
        return p;
    };
    const size_t NC  = (size_t)NPAD * CDIM;
    const size_t NC3 = (size_t)NPAD * C3;

    float*  h     = (float*)carve(NC * 4);
    float*  m     = (float*)carve(NC * 4);
    float*  total = (float*)carve(NC * 4);
    float*  xbuf  = (float*)carve(NC * 4);   // conv output; later head1 output
    float*  agg   = (float*)carve(NC * 4);
    float*  gi    = (float*)carve(NC3 * 4);
    float*  gh    = (float*)carve(NC3 * 4);
    __bf16* wconv = (__bf16*)carve((size_t)NBLOCKS * NET * NSTEPS * CDIM * CDIM * 2);
    __bf16* wih   = (__bf16*)carve((size_t)NBLOCKS * NET * CDIM * C3 * 2);
    __bf16* whh   = (__bf16*)carve((size_t)NBLOCKS * NET * CDIM * C3 * 2);
    __bf16* w1b   = (__bf16*)carve((size_t)CDIM * CDIM * 2);

    const int T256 = 256;
    auto blk = [](int n) { return (n + 255) / 256; };

    // -------- weight preparation: bf16, pre-swizzled WMMA-B fragments ------
    {
        const int nconv = NBLOCKS * NET * NSTEPS * CDIM * CDIM;   // 18 mats, K x ncols
        repack_b<<<blk(nconv), T256, 0, stream>>>(conv_w, wconv, CDIM, 0, nconv);
        const int ngru = NBLOCKS * NET * C3 * CDIM;               // 6 mats, W[j][k] -> B[k][j]
        repack_b<<<blk(ngru), T256, 0, stream>>>(gwih, wih, C3, 1, ngru);
        repack_b<<<blk(ngru), T256, 0, stream>>>(gwhh, whh, C3, 1, ngru);
        const int nh1 = CDIM * CDIM;
        repack_b<<<blk(nh1), T256, 0, stream>>>(hw1, w1b, CDIM, 1, nh1);
    }

    // -------- feature build (zero-padded rows) --------
    build_features<<<blk((int)NC), T256, 0, stream>>>(x_type, x_tok, x_small, h);

    auto gemm = [&](const float* A, const __bf16* B, const float* bias,
                    float* D, int ncols, int relu) {
        dim3 g(NPAD / 64, ncols / 128);
        gemm_wmma_bf16<<<g, 128, 0, stream>>>(A, B, bias, D, ncols, relu);
    };

    // -------- GGNN blocks --------
    for (int b = 0; b < NBLOCKS; ++b) {
        zero_f32<<<blk((int)NC), T256, 0, stream>>>(total, (int)NC);
        for (int t = 0; t < NET; ++t) {
            copy_f32<<<blk((int)NC), T256, 0, stream>>>(m, h, (int)NC);
            const int bt = b * NET + t;
            for (int s = 0; s < NSTEPS; ++s) {
                // x = m @ conv_w[b,t,s]
                gemm(m, wconv + (size_t)(bt * NSTEPS + s) * CDIM * CDIM,
                     nullptr, xbuf, CDIM, 0);
                // agg = segment_sum(x[src] * mask(t), dst)
                zero_f32<<<blk((int)NC), T256, 0, stream>>>(agg, (int)NC);
                scatter_edges<<<NEDGES / 8, T256, 0, stream>>>(xbuf, esrc, edst,
                                                               etype, t, agg);
                // gi = agg @ wih^T + bih ; gh = m @ whh^T + bhh
                gemm(agg, wih + (size_t)bt * CDIM * C3, gbih + (size_t)bt * C3,
                     gi, C3, 0);
                gemm(m,   whh + (size_t)bt * CDIM * C3, gbhh + (size_t)bt * C3,
                     gh, C3, 0);
                // m = GRUCell(agg, m)
                gru_eltwise<<<blk((int)NC), T256, 0, stream>>>(gi, gh, m);
            }
            add_f32<<<blk((int)NC), T256, 0, stream>>>(total, m, (int)NC);
        }
        // h = relu(LayerNorm(h + total))
        ln_relu<<<NPAD / 8, T256, 0, stream>>>(total, ln_g + b * CDIM,
                                               ln_b + b * CDIM, h);
    }

    // -------- head --------
    gemm(h, w1b, hb1, xbuf, CDIM, 1);                        // z = relu(h@w1^T+b1)
    head2_kernel<<<(NNODES + 7) / 8, T256, 0, stream>>>(xbuf, hw2, hb2,
                                                        (float*)d_out);
}
